// DiscriminationLoss_61409442398556
// MI455X (gfx1250) — matrix operand order repaired and checked by reading, not compiled, not run on hardware
//
#include <hip/hip_runtime.h>

typedef __attribute__((ext_vector_type(16))) __bf16 v16bf;
typedef __attribute__((ext_vector_type(8)))  float  v8f;

#define IMG_DIM 640
#define HW      (IMG_DIM * IMG_DIM)   // 409600 pixels per image
#define NB      16                    // batch
#define NCH     4                     // channels
#define BLOCKS_X 100
#define WAVES_PER_BLOCK 8
#define PX_PER_WAVE (HW / (BLOCKS_X * WAVES_PER_BLOCK))  // 512

// Workspace G: per image a 16x16 f32 matrix mirroring the WMMA D layout:
//   rows m=0..15  <-> label m+1
//   cols 0..3  = bf16-hi partial channel sums
//   col  4     = pixel counts for label m+1
//   cols 5..8  = bf16-lo residual partial channel sums
// G size = 16 * 256 floats = 16 KB.

__global__ void disc_zero(float* __restrict__ G) {
    G[blockIdx.x * 256 + threadIdx.x] = 0.0f;
}

__global__ __launch_bounds__(256) void disc_pass1(
    const float* __restrict__ pred,     // [B][C][H][W]
    const int*   __restrict__ labels,   // [B][H][W]
    float*       __restrict__ G)        // [B][16][16]
{
    __shared__ float blk[256];
    const int tid = threadIdx.x;
    blk[tid] = 0.0f;
    __syncthreads();

    const int b    = blockIdx.y;
    const int lane = tid & 31;
    const int wave = tid >> 5;
    const int n    = lane & 15;   // B/D column; also A row (label id n+1)
    const int h    = lane >> 4;   // lane half (selects K range of fragments)

    const int wave_glob = blockIdx.x * WAVES_PER_BLOCK + wave;   // 0..799
    const int start     = wave_glob * PX_PER_WAVE;

    const float* predb = pred   + (size_t)b * NCH * HW;
    const int*   labb  = labels + (size_t)b * HW;

    // Loop-invariant lane classification for the B fragment columns
    const int  cc     = (n < 4) ? n : ((n >= 5 && n <= 8) ? (n - 5) : 0);
    const bool isHi   = (n < 4);
    const bool isLo   = (n >= 5 && n <= 8);
    const bool isOne  = (n == 4);
    const bool isData = isHi || isLo;
    const int  target = n + 1;    // A one-hot: row m==n matches label n+1

    // Preheader-initialized B fragment: lanes with constant columns
    // (col 4 = ones, cols 9..15 = zeros) never touch bm inside the loop.
    v16bf bm;
#pragma unroll
    for (int i = 0; i < 16; ++i) bm[i] = isOne ? (__bf16)1.0f : (__bf16)0.0f;

    v8f acc = {0.f, 0.f, 0.f, 0.f, 0.f, 0.f, 0.f, 0.f};

    for (int g0 = 0; g0 < PX_PER_WAVE; g0 += 32) {
        const int base = start + g0;   // 32-pixel group, 64B aligned

        // ---- B fragment (32x16 bf16): lane holds col n, K = h*16 + 0..15 ----
        if (isData) {
            const float* vp = predb + (size_t)cc * HW + base + h * 16;
            const float4 x0 = *(const float4*)(vp + 0);
            const float4 x1 = *(const float4*)(vp + 4);
            const float4 x2 = *(const float4*)(vp + 8);
            const float4 x3 = *(const float4*)(vp + 12);
            float xs[16];
            xs[0]=x0.x;  xs[1]=x0.y;  xs[2]=x0.z;  xs[3]=x0.w;
            xs[4]=x1.x;  xs[5]=x1.y;  xs[6]=x1.z;  xs[7]=x1.w;
            xs[8]=x2.x;  xs[9]=x2.y;  xs[10]=x2.z; xs[11]=x2.w;
            xs[12]=x3.x; xs[13]=x3.y; xs[14]=x3.z; xs[15]=x3.w;
            if (isHi) {
                // 8x v_cvt_pk_bf16_f32
#pragma unroll
                for (int i = 0; i < 16; ++i) bm[i] = (__bf16)xs[i];
            } else {
                // split-bf16 residual: lo = bf16(x - f32(bf16(x)))
#pragma unroll
                for (int i = 0; i < 16; ++i) {
                    const __bf16 hi = (__bf16)xs[i];
                    bm[i] = (__bf16)(xs[i] - (float)hi);
                }
            }
        }

        // ---- A fragment (16x32 bf16 one-hot): lane row n,
        //      K = h*16 + 0..15 (matches B fragment) ----
        const int* lp = labb + base + h * 16;
        const int4 l0 = *(const int4*)(lp + 0);
        const int4 l1 = *(const int4*)(lp + 4);
        const int4 l2 = *(const int4*)(lp + 8);
        const int4 l3 = *(const int4*)(lp + 12);
        int ls[16];
        ls[0]=l0.x;  ls[1]=l0.y;  ls[2]=l0.z;  ls[3]=l0.w;
        ls[4]=l1.x;  ls[5]=l1.y;  ls[6]=l1.z;  ls[7]=l1.w;
        ls[8]=l2.x;  ls[9]=l2.y;  ls[10]=l2.z; ls[11]=l2.w;
        ls[12]=l3.x; ls[13]=l3.y; ls[14]=l3.z; ls[15]=l3.w;
        v16bf am;
#pragma unroll
        for (int i = 0; i < 16; ++i)
            am[i] = (ls[i] == target) ? (__bf16)1.0f : (__bf16)0.0f;

        // D[m][c] += one-hot x values : channel sums + counts in one op
        acc = __builtin_amdgcn_wmma_f32_16x16x32_bf16(
            /*neg_a=*/false, am, /*neg_b=*/false, bm,
            /*c_mod=*/(short)0, acc, /*reuse_a=*/false, /*reuse_b=*/false);
    }

    // D layout: lane holds col n, rows v + 8*h for VGPR v
    if (n <= 8) {
#pragma unroll
        for (int v = 0; v < 8; ++v) {
            const int row = v + 8 * h;
            atomicAdd(&blk[row * 16 + n], acc[v]);
        }
    }
    __syncthreads();

    if ((tid & 15) <= 8) {
        atomicAdd(&G[(size_t)b * 256 + tid], blk[tid]);
    }
}

__global__ void disc_finalize(const float* __restrict__ G, float* __restrict__ out) {
    __shared__ double shP[NB], shOwn[NB], shKb[NB];
    const int lane = threadIdx.x;
    if (lane < NB) {
        const float* g = G + lane * 256;
        double sum_g = 0.0, cnt_sum = 0.0, kb = 0.0;
        for (int k = 0; k < 16; ++k) {
            const double s0 = (double)g[k*16+0] + (double)g[k*16+5];
            const double s1 = (double)g[k*16+1] + (double)g[k*16+6];
            const double s2 = (double)g[k*16+2] + (double)g[k*16+7];
            const double s3 = (double)g[k*16+3] + (double)g[k*16+8];
            const double cnt = (double)g[k*16+4];
            const double N = sqrt(s0*s0 + s1*s1 + s2*s2 + s3*s3);
            double r = 3.0 - N; r = (r > 0.0) ? r : 0.0;
            const double f = log(r*r + 1.0);
            sum_g  += cnt * f;
            cnt_sum += cnt;
            if (cnt > 0.0) kb = (double)(k + 1);   // max label present
        }
        const double F0   = log(10.0);             // log(sigma^2 + 1)
        const double cnt0 = (double)HW - cnt_sum;  // background pixels
        sum_g += cnt0 * F0;                        // label-0 pixels gather F0
        const bool   active = (kb > 1.0);
        const double P = kb * (kb - 1.0) * 0.5;
        shOwn[lane] = active ? ((kb - 1.0) * sum_g + (double)HW * (P - (kb - 1.0)) * F0)
                             : 0.0;
        shP[lane]   = active ? P : 0.0;
        shKb[lane]  = kb;
    }
    __syncthreads();
    if (lane == 0) {
        const double F0 = log(10.0);
        double sumP = 0.0;
        for (int b = 0; b < NB; ++b) sumP += shP[b];
        double res = 0.0;
        for (int b = 0; b < NB; ++b) {
            const double kb    = shKb[b];
            const double other = (sumP - shP[b]) * (double)HW * F0;
            const double scale = (kb > 1.0) ? 1.0 / (kb * (kb - 1.0)) : kb;
            res += scale * (shOwn[b] + other);
        }
        out[0] = (float)res;
    }
}

extern "C" void kernel_launch(void* const* d_in, const int* in_sizes, int n_in,
                              void* d_out, int out_size, void* d_ws, size_t ws_size,
                              hipStream_t stream) {
    const float* pred   = (const float*)d_in[0];
    const int*   labels = (const int*)d_in[1];
    float*       out    = (float*)d_out;
    float*       G      = (float*)d_ws;   // 16 KB accumulator

    disc_zero<<<NB, 256, 0, stream>>>(G);
    disc_pass1<<<dim3(BLOCKS_X, NB), 256, 0, stream>>>(pred, labels, G);
    disc_finalize<<<1, 32, 0, stream>>>(G, out);
}
